// CoDiscriminator_88648124991515
// MI455X (gfx1250) — matrix operand order repaired and checked by reading, not compiled, not run on hardware
//
#include <hip/hip_runtime.h>
#include <cmath>
#include <cstdint>

#define DEV static __device__ __forceinline__

typedef _Float16 half_t;
typedef __attribute__((ext_vector_type(16))) _Float16 v16h;
typedef __attribute__((ext_vector_type(8)))  _Float16 v8h;
typedef __attribute__((ext_vector_type(8)))  float    v8f;

// Problem constants
static constexpr int B_ = 32, S_ = 1024, D_ = 512, K_ = 256, H_ = 512;

// GEMM tiling: 256 threads (8 wave32), block tile 128x256, K-step 32 (one WMMA depth),
// wave tile 64x64 (16 WMMAs per K-step per wave, 1:1 wmma:ds_load_b128 ratio),
// double-buffered LDS (one barrier per K-step).
static constexpr int TM = 128, TN = 256, TKK = 32;
static constexpr int LDK = TKK + 8;  // LDS row pad (f16 elems) -> 80B row stride

// ---- CDNA5 async global->LDS path (guarded; falls back to staged copies) ----
#if defined(__AMDGCN__) && \
    __has_builtin(__builtin_amdgcn_global_load_async_to_lds_b128) && \
    __has_builtin(__builtin_amdgcn_s_wait_asynccnt)
#define ASYNC_LDS 1
typedef __attribute__((__vector_size__(4 * sizeof(int)))) int vi4;
typedef __attribute__((address_space(1))) vi4* as1_v4p;   // HIP prints AS1 as __device__
typedef __attribute__((address_space(3))) vi4* as3_v4p;
DEV void async_cp16(const void* g, void* l) {
  __builtin_amdgcn_global_load_async_to_lds_b128(
      (as1_v4p)(uintptr_t)g, (as3_v4p)(uintptr_t)l, 0, 0);
}
DEV void async_wait0() { __builtin_amdgcn_s_wait_asynccnt(0); }
#else
#define ASYNC_LDS 0
DEV void async_cp16(const void*, void*) {}
DEV void async_wait0() {}
#endif

DEV void load16(const float* s, half_t* d) {
  const float4* f = (const float4*)s;
#pragma unroll
  for (int i = 0; i < 4; ++i) {
    float4 v = f[i];
    d[4*i+0] = (half_t)v.x; d[4*i+1] = (half_t)v.y;
    d[4*i+2] = (half_t)v.z; d[4*i+3] = (half_t)v.w;
  }
}
DEV void load16(const half_t* s, half_t* d) {
  const v8h* h = (const v8h*)s;
  v8h a = h[0], b = h[1];
  *(v8h*)d = a;
  *((v8h*)d + 1) = b;
}

// C[M,N](f32 acc) = A[M,K] * B[K,N], batched (stride 0 => shared operand).
// !TRA: Ag row-major [M,K]   |  TRA: Ag row-major [K,M] (i.e. A = Ag^T)
// !TRB: Bg row-major [K,N]   |  TRB: Bg row-major [N,K] (i.e. B = Bg^T)
template <typename TA, typename TB, bool TRA, bool TRB, bool OUT_HALF, bool DO_TANH>
__global__ __launch_bounds__(256) void gemm_wmma(
    const TA* __restrict__ Ag, long lda, long sA,
    const TB* __restrict__ Bg, long ldb, long sB,
    void* __restrict__ Cg, long ldc, long sC,
    int M, int N, int Kd)
{
  __shared__ __align__(16) half_t As[2][TM][LDK];
  __shared__ __align__(16) half_t Bs[2][TN][LDK];   // stored transposed: Bs[n][k]

  constexpr bool ASYNC_A = ASYNC_LDS && (sizeof(TA) == 2) && !TRA;
  constexpr bool ASYNC_B = ASYNC_LDS && (sizeof(TB) == 2) && TRB;
  constexpr bool ANY_ASYNC = ASYNC_A || ASYNC_B;

  const int tid = threadIdx.x;
  const int m0 = blockIdx.y * TM;
  const int n0 = blockIdx.x * TN;
  Ag += (size_t)blockIdx.z * sA;
  Bg += (size_t)blockIdx.z * sB;

  const int wid = tid >> 5, lane = tid & 31;
  const int wm = (wid & 1) * 64;   // 2 waves along M, 64 rows each
  const int wn = (wid >> 1) * 64;  // 4 waves along N, 64 cols each
  const int lr = lane & 15, lh = lane >> 4;

  v8f acc[4][4];
#pragma unroll
  for (int ms = 0; ms < 4; ++ms)
#pragma unroll
    for (int ns = 0; ns < 4; ++ns)
#pragma unroll
      for (int r = 0; r < 8; ++r) acc[ms][ns][r] = 0.f;

  // ---- tile stagers ----
  auto loadA = [&](int buf, int kk) {
    if (!TRA) {
      int r = tid >> 1, seg = (tid & 1) * 16;
      const TA* src = Ag + (size_t)(m0 + r) * lda + kk + seg;
      half_t* dst = &As[buf][r][seg];
      if constexpr (sizeof(TA) == 2) {
        if constexpr (ASYNC_A) {
          async_cp16(src, dst);
          async_cp16((const char*)src + 16, dst + 8);
        } else {
          load16((const half_t*)src, dst);
        }
      } else {
        load16((const float*)src, dst);
      }
    } else {
      int k = tid >> 3, mseg = (tid & 7) * 16;
      const TA* src = Ag + (size_t)(kk + k) * lda + m0 + mseg;
#pragma unroll
      for (int i = 0; i < 16; ++i) As[buf][mseg + i][k] = (half_t)src[i];
      if (kk + TKK < Kd)
        __builtin_prefetch(src + (size_t)TKK * lda, 0, 1);
    }
  };
  auto loadB = [&](int buf, int kk) {
    if (TRB) {
      int n = tid;  // 0..255, one full 32-elem row per thread
      const TB* src = Bg + (size_t)(n0 + n) * ldb + kk;
      half_t* dst = &Bs[buf][n][0];
      if constexpr (sizeof(TB) == 2) {
        if constexpr (ASYNC_B) {
          async_cp16(src, dst);
          async_cp16((const char*)src + 16, dst + 8);
          async_cp16((const char*)src + 32, dst + 16);
          async_cp16((const char*)src + 48, dst + 24);
        } else {
          load16((const half_t*)src, dst);
          load16((const half_t*)src + 16, dst + 16);
        }
      } else {
        load16((const float*)src, dst);
        load16((const float*)src + 16, dst + 16);
      }
    } else {
      int k = tid >> 3, nseg = (tid & 7) * 32;
      const TB* src = Bg + (size_t)(kk + k) * ldb + n0 + nseg;
#pragma unroll
      for (int i = 0; i < 32; ++i) Bs[buf][nseg + i][k] = (half_t)src[i];
      if (kk + TKK < Kd)
        __builtin_prefetch(src + (size_t)TKK * ldb, 0, 1);
    }
  };

  // ---- prologue: fill buffer 0 ----
  loadA(0, 0);
  loadB(0, 0);
  if constexpr (ANY_ASYNC) async_wait0();
  __syncthreads();

  int p = 0;
  for (int kk = 0; kk < Kd; kk += TKK) {
    const int kn = kk + TKK;
    if (kn < Kd) {          // stage next tile into the other buffer
      loadA(p ^ 1, kn);
      loadB(p ^ 1, kn);
    }

    // ---- compute on buffer p (CDNA5 16-bit WMMA fragment layouts)
    // A 16x32: lane<16 -> M=lane, K in {0..7,16..23}; lane>=16 -> K in {8..15,24..31}
    // B 32x16 (as Bs[n][k]): N = lane%16, K same as A: lane<16 -> {0..7,16..23}; lane>=16 -> {8..15,24..31}
    v16h afr[4];
#pragma unroll
    for (int ms = 0; ms < 4; ++ms) {
      int row = wm + ms * 16 + lr;
      v8h lo = *(const v8h*)&As[p][row][lh * 8];
      v8h hi = *(const v8h*)&As[p][row][lh * 8 + 16];
      afr[ms] = __builtin_shufflevector(lo, hi, 0,1,2,3,4,5,6,7,8,9,10,11,12,13,14,15);
    }
#pragma unroll
    for (int ns = 0; ns < 4; ++ns) {
      int n = wn + ns * 16 + lr;
      v8h lo = *(const v8h*)&Bs[p][n][lh * 8];
      v8h hi = *(const v8h*)&Bs[p][n][lh * 8 + 16];
      v16h bfr = __builtin_shufflevector(lo, hi, 0,1,2,3,4,5,6,7,8,9,10,11,12,13,14,15);
#pragma unroll
      for (int ms = 0; ms < 4; ++ms)
        acc[ms][ns] = __builtin_amdgcn_wmma_f32_16x16x32_f16(
            false, afr[ms], false, bfr, (short)0, acc[ms][ns], false, false);
    }

    if constexpr (ANY_ASYNC) {
      if (kn < Kd) async_wait0();   // next buffer's DMA done before barrier
    }
    __syncthreads();                // staged stores (dscnt) handled by compiler
    p ^= 1;
  }

  // ---- epilogue: C/D f32 16x16 layout: VGPR r -> M = r + (lane/16)*8, N = lane%16
  half_t* Ch = (half_t*)Cg;
  float*  Cf = (float*)Cg;
  const size_t cb = (size_t)blockIdx.z * sC;
#pragma unroll
  for (int ms = 0; ms < 4; ++ms)
#pragma unroll
    for (int ns = 0; ns < 4; ++ns) {
#pragma unroll
      for (int r = 0; r < 8; ++r) {
        float v = acc[ms][ns][r];
        if (DO_TANH) v = tanhf(v);
        size_t m = (size_t)(m0 + wm + ms * 16 + r + lh * 8);
        size_t n = (size_t)(n0 + wn + ns * 16 + lr);
        if (OUT_HALF) Ch[cb + m * (size_t)ldc + n] = (half_t)v;
        else          Cf[cb + m * (size_t)ldc + n] = v;
      }
    }
}

// ---- DFT basis generation (symmetric cos/sin matrices, f16) ----
__global__ void gen_dft_d(half_t* Cd, half_t* Sd) {
  int j = blockIdx.x * 256 + threadIdx.x;  // 0..511
  int i = blockIdx.y;                      // 0..511
  int prod = (i * j) & (D_ - 1);           // mod D (power of two)
  float ang = (float)prod * (6.28318530717958647692f / (float)D_);
  float s, c;
  __sincosf(ang, &s, &c);
  Cd[(size_t)i * D_ + j] = (half_t)c;
  Sd[(size_t)i * D_ + j] = (half_t)s;
}

__global__ void gen_cs2(half_t* CS2) {   // CS2 = [Cs | -Ss], [S, 2S]
  int j = blockIdx.x * 256 + threadIdx.x;  // 0..2047
  int i = blockIdx.y;                      // 0..1023
  int jj = j & (S_ - 1);
  int prod = (i * jj) & (S_ - 1);
  float ang = (float)prod * (6.28318530717958647692f / (float)S_);
  float s, c;
  __sincosf(ang, &s, &c);
  CS2[(size_t)i * (2 * S_) + j] = (half_t)((j < S_) ? c : -s);
}

// logit[row] = sum_k w[k] * tanh(P[row,k] + G[row,k]),  K_=256, one block per row
__global__ __launch_bounds__(256) void logit_kernel(const half_t* __restrict__ Pm,
                                                    const float* __restrict__ Gm,
                                                    const float* __restrict__ w,
                                                    float* __restrict__ out) {
  int row = blockIdx.x;
  int k = threadIdx.x;
  __shared__ float red[256];
  size_t idx = (size_t)row * K_ + k;
  red[k] = w[k] * tanhf((float)Pm[idx] + Gm[idx]);
  __syncthreads();
  for (int off = 128; off > 0; off >>= 1) {
    if (k < off) red[k] += red[k + off];
    __syncthreads();
  }
  if (k == 0) out[row] = red[0];
}

__global__ __launch_bounds__(256) void softmax_kernel(const float* __restrict__ x,
                                                      float* __restrict__ a) {
  int b = blockIdx.x, t = threadIdx.x;
  __shared__ float red[256];
  const float* xb = x + (size_t)b * S_;
  float m = -3.402823466e38f;
  for (int i = t; i < S_; i += 256) m = fmaxf(m, xb[i]);
  red[t] = m; __syncthreads();
  for (int off = 128; off > 0; off >>= 1) {
    if (t < off) red[t] = fmaxf(red[t], red[t + off]);
    __syncthreads();
  }
  m = red[0]; __syncthreads();
  float s = 0.f;
  for (int i = t; i < S_; i += 256) s += expf(xb[i] - m);
  red[t] = s; __syncthreads();
  for (int off = 128; off > 0; off >>= 1) {
    if (t < off) red[t] += red[t + off];
    __syncthreads();
  }
  float inv = 1.f / red[0];
  for (int i = t; i < S_; i += 256) a[(size_t)b * S_ + i] = expf(xb[i] - m) * inv;
}

// co[b, 0:512] = sum_s Ai[b,s]*img[b,s,:]; co[b, 512:1024] = sum_s Ac[b,s]*cap[b,s,:]
__global__ __launch_bounds__(256) void attend_kernel(const float* __restrict__ Ai,
                                                     const float* __restrict__ Ac,
                                                     const float* __restrict__ img,
                                                     const float* __restrict__ cap,
                                                     float* __restrict__ co) {
  int b = blockIdx.y;
  int d = blockIdx.x * 256 + threadIdx.x;  // 0..511
  float as = 0.f, ac = 0.f;
  for (int s = 0; s < S_; ++s) {
    float wi = Ai[(size_t)b * S_ + s];
    float wc = Ac[(size_t)b * S_ + s];
    as += wi * img[((size_t)b * S_ + s) * D_ + d];
    ac += wc * cap[((size_t)b * S_ + s) * D_ + d];
  }
  co[(size_t)b * (2 * D_) + d]      = as;
  co[(size_t)b * (2 * D_) + D_ + d] = ac;
}

// out[b] = sigmoid( relu(co[b,:]@W1 + b1) @ W2 + b2 ), one block per batch
__global__ __launch_bounds__(256) void mlp_kernel(const float* __restrict__ co,
                                                  const float* __restrict__ W1,
                                                  const float* __restrict__ b1,
                                                  const float* __restrict__ W2,
                                                  const float* __restrict__ b2,
                                                  float* __restrict__ out) {
  int b = blockIdx.x, t = threadIdx.x;
  __shared__ float sco[2 * D_];
  __shared__ float red[256];
  for (int i = t; i < 2 * D_; i += 256) sco[i] = co[(size_t)b * (2 * D_) + i];
  __syncthreads();
  float part = 0.f;
  for (int h = t; h < H_; h += 256) {
    float acc = b1[h];
    for (int j = 0; j < 2 * D_; ++j) acc += sco[j] * W1[(size_t)j * H_ + h];
    part += fmaxf(acc, 0.f) * W2[h];
  }
  red[t] = part; __syncthreads();
  for (int off = 128; off > 0; off >>= 1) {
    if (t < off) red[t] += red[t + off];
    __syncthreads();
  }
  if (t == 0) out[b] = 1.f / (1.f + expf(-(red[0] + b2[0])));
}

extern "C" void kernel_launch(void* const* d_in, const int* in_sizes, int n_in,
                              void* d_out, int out_size, void* d_ws, size_t ws_size,
                              hipStream_t stream) {
  (void)in_sizes; (void)n_in; (void)out_size; (void)ws_size;
  const float* img = (const float*)d_in[0];
  const float* cap = (const float*)d_in[1];
  const float* Wl  = (const float*)d_in[2];
  const float* Wc  = (const float*)d_in[3];
  const float* Wi  = (const float*)d_in[4];
  const float* wHi = (const float*)d_in[5];
  const float* whc = (const float*)d_in[6];
  const float* W1  = (const float*)d_in[7];
  const float* b1  = (const float*)d_in[8];
  const float* W2  = (const float*)d_in[9];
  const float* b2  = (const float*)d_in[10];
  float* out = (float*)d_out;

  char* p = (char*)d_ws;
  auto alloc = [&](size_t bytes) -> char* {
    char* r = p;
    p += (bytes + 255) & ~(size_t)255;
    return r;
  };
  const long SD = (long)S_ * D_;   // 524288
  const long SS = (long)S_ * S_;
  const long SK = (long)S_ * K_;

  half_t* Cd   = (half_t*)alloc((size_t)D_ * D_ * 2);
  half_t* Sd   = (half_t*)alloc((size_t)D_ * D_ * 2);
  half_t* CS2  = (half_t*)alloc((size_t)S_ * 2 * S_ * 2);
  half_t* RI   = (half_t*)alloc((size_t)B_ * 2 * SD * 2);  // [B,2S,D]; reused as T and G
  half_t* imgf = (half_t*)alloc((size_t)B_ * SD * 2);
  half_t* capf = (half_t*)alloc((size_t)B_ * SD * 2);
  half_t* L    = (half_t*)alloc((size_t)B_ * SS * 2);
  half_t* Pb   = (half_t*)alloc((size_t)B_ * SK * 2);
  half_t* Qb   = (half_t*)alloc((size_t)B_ * SK * 2);
  float* logit_i = (float*)alloc((size_t)B_ * S_ * 4);
  float* logit_c = (float*)alloc((size_t)B_ * S_ * 4);
  float* Ai = (float*)alloc((size_t)B_ * S_ * 4);
  float* Ac = (float*)alloc((size_t)B_ * S_ * 4);
  float* co = (float*)alloc((size_t)B_ * 2 * D_ * 4);
  half_t* T = RI;                                   // alias: RI dead after cap_f
  float*  G = (float*)(RI + (size_t)B_ * SD);       // alias: upper half of RI

  dim3 blk(256);

  gen_dft_d<<<dim3(2, 512), blk, 0, stream>>>(Cd, Sd);
  gen_cs2<<<dim3(8, 1024), blk, 0, stream>>>(CS2);

  // ---- img_f = CS2 @ [img@Cd ; img@Sd]   (fft2(img).real)
  gemm_wmma<float, half_t, false, false, true, false>
      <<<dim3(D_/TN, S_/TM, B_), blk, 0, stream>>>(img, D_, SD, Cd, D_, 0,
                                                   RI, D_, 2*SD, S_, D_, D_);
  gemm_wmma<float, half_t, false, false, true, false>
      <<<dim3(D_/TN, S_/TM, B_), blk, 0, stream>>>(img, D_, SD, Sd, D_, 0,
                                                   RI + SD, D_, 2*SD, S_, D_, D_);
  gemm_wmma<half_t, half_t, false, false, true, false>
      <<<dim3(D_/TN, S_/TM, B_), blk, 0, stream>>>(CS2, 2*S_, 0, RI, D_, 2*SD,
                                                   imgf, D_, SD, S_, D_, 2*S_);
  // ---- cap_f
  gemm_wmma<float, half_t, false, false, true, false>
      <<<dim3(D_/TN, S_/TM, B_), blk, 0, stream>>>(cap, D_, SD, Cd, D_, 0,
                                                   RI, D_, 2*SD, S_, D_, D_);
  gemm_wmma<float, half_t, false, false, true, false>
      <<<dim3(D_/TN, S_/TM, B_), blk, 0, stream>>>(cap, D_, SD, Sd, D_, 0,
                                                   RI + SD, D_, 2*SD, S_, D_, D_);
  gemm_wmma<half_t, half_t, false, false, true, false>
      <<<dim3(D_/TN, S_/TM, B_), blk, 0, stream>>>(CS2, 2*S_, 0, RI, D_, 2*SD,
                                                   capf, D_, SD, S_, D_, 2*S_);

  // ---- T = img_f @ Wl ; L = tanh(T @ img_f^T)
  gemm_wmma<half_t, float, false, false, true, false>
      <<<dim3(D_/TN, S_/TM, B_), blk, 0, stream>>>(imgf, D_, SD, Wl, D_, 0,
                                                   T, D_, SD, S_, D_, D_);
  gemm_wmma<half_t, half_t, false, true, true, true>
      <<<dim3(S_/TN, S_/TM, B_), blk, 0, stream>>>(T, D_, SD, imgf, D_, SD,
                                                   L, S_, SS, S_, S_, D_);

  // ---- P = img_f @ Wi^T ; Q = cap_f @ Wc^T   ([B,S,K])
  gemm_wmma<half_t, float, false, true, true, false>
      <<<dim3(K_/TN, S_/TM, B_), blk, 0, stream>>>(imgf, D_, SD, Wi, D_, 0,
                                                   Pb, K_, SK, S_, K_, D_);
  gemm_wmma<half_t, float, false, true, true, false>
      <<<dim3(K_/TN, S_/TM, B_), blk, 0, stream>>>(capf, D_, SD, Wc, D_, 0,
                                                   Qb, K_, SK, S_, K_, D_);

  // ---- Hi^T = tanh(P + L^T @ Q): logit_i[b,s] = sum_k wHi[k]*Hi^T[s,k]
  gemm_wmma<half_t, half_t, true, false, false, false>
      <<<dim3(K_/TN, S_/TM, B_), blk, 0, stream>>>(L, S_, SS, Qb, K_, SK,
                                                   G, K_, SK, S_, K_, S_);
  logit_kernel<<<B_ * S_, blk, 0, stream>>>(Pb, G, wHi, logit_i);

  // ---- Hc^T = tanh(Q + L @ P): logit_c
  gemm_wmma<half_t, half_t, false, false, false, false>
      <<<dim3(K_/TN, S_/TM, B_), blk, 0, stream>>>(L, S_, SS, Pb, K_, SK,
                                                   G, K_, SK, S_, K_, S_);
  logit_kernel<<<B_ * S_, blk, 0, stream>>>(Qb, G, whc, logit_c);

  // ---- softmax, attention over ORIGINAL reps, MLP head
  softmax_kernel<<<B_, blk, 0, stream>>>(logit_i, Ai);
  softmax_kernel<<<B_, blk, 0, stream>>>(logit_c, Ac);
  attend_kernel<<<dim3(2, B_), blk, 0, stream>>>(Ai, Ac, img, cap, co);
  mlp_kernel<<<B_, blk, 0, stream>>>(co, W1, b1, W2, b2, out);
}
